// RetBlock_29248727286041
// MI455X (gfx1250) — matrix operand order repaired and checked by reading, not compiled, not run on hardware
//
#include <hip/hip_runtime.h>

// ---------------------------------------------------------------------------
// RetNet retention block for MI455X (gfx1250, wave32, WMMA bf16 16x16x32,
// Tensor Data Mover double-buffering for the streaming k-tiles).
//   B=4, L=4096, HIDDEN=2048, HEAD=256, GAMMA=0.96875, SCALE_BASE=512
// ---------------------------------------------------------------------------

#define BATCH   4
#define LSEQ    4096
#define HID     2048
#define HD      256

typedef __attribute__((ext_vector_type(16))) __bf16    v16bf;
typedef __attribute__((ext_vector_type(8)))  __bf16    v8bf;
typedef __attribute__((ext_vector_type(8)))  float     v8f;
typedef __attribute__((ext_vector_type(4)))  unsigned  u32x4;
typedef __attribute__((ext_vector_type(4)))  int       i32x4;
typedef __attribute__((ext_vector_type(8)))  int       i32x8;

__device__ __forceinline__ v8f wmma_bf16(v16bf a, v16bf b, v8f c) {
    // D = A(16x32 bf16) * B(32x16 bf16) + C(16x16 f32)
    return __builtin_amdgcn_wmma_f32_16x16x32_bf16(
        /*neg_a=*/false, a, /*neg_b=*/false, b,
        /*c_mod=*/(short)0, c, /*reuse_a=*/false, /*reuse_b=*/false);
}

// ---------------------------------------------------------------------------
// TDM: issue a 2-D tile load Global -> LDS (tensor_load_to_lds, TENSORcnt).
// D# per CDNA5 ISA:
//   group0: count=1 | lds_addr | global_addr[56:0] | type=2
//   group1: data_size=2B, tensor_dim0/1, tile_dim0/1, tensor_dim0_stride
//   groups 2/3: zero (tile_dim2/3 unused -> 2-D)
// ---------------------------------------------------------------------------
__device__ __forceinline__ void tdm_load_tile_2d(const void* gptr,
                                                 unsigned lds_byte_off,
                                                 unsigned tile_d0, unsigned tile_d1,
                                                 unsigned tens_d0, unsigned tens_d1,
                                                 unsigned stride0) {
    unsigned long long ga = (unsigned long long)gptr;
    u32x4 g0 = { 1u,                                   // count=1, user mode
                 lds_byte_off,                         // lds_addr (bytes)
                 (unsigned)ga,                         // global_addr[31:0]
                 (unsigned)(ga >> 32) | (2u << 30) };  // addr[56:32] | type=2
    i32x8 g1 = { (int)(1u << 16),                                       // data_size=1 (2B)
                 (int)((tens_d0 & 0xFFFFu) << 16),                      // tensor_dim0 lo
                 (int)(((tens_d0 >> 16) & 0xFFFFu) |
                       ((tens_d1 & 0xFFFFu) << 16)),                    // dim0 hi | dim1 lo
                 (int)(((tens_d1 >> 16) & 0xFFFFu) |
                       ((tile_d0 & 0xFFFFu) << 16)),                    // dim1 hi | tile_dim0
                 (int)(tile_d1 & 0xFFFFu),                              // tile_dim1 (tile_dim2=0)
                 (int)stride0,                                          // dim0_stride[31:0]
                 0, 0 };
    i32x4 gz = { 0, 0, 0, 0 };
#if defined(__clang_major__) && (__clang_major__ >= 23)
    i32x8 gz8 = { 0, 0, 0, 0, 0, 0, 0, 0 };
    __builtin_amdgcn_tensor_load_to_lds(g0, g1, gz, gz, gz8, 0);
#else
    __builtin_amdgcn_tensor_load_to_lds(g0, g1, gz, gz, 0);
#endif
}

// ---------------------------------------------------------------------------
// Kernel 1: W (2048x256 f32, row-major) -> W^T (256x2048 bf16).
// ---------------------------------------------------------------------------
__global__ void wt_kernel(const float* __restrict__ wq,
                          const float* __restrict__ wk,
                          const float* __restrict__ wv,
                          __bf16* __restrict__ oq,
                          __bf16* __restrict__ ok,
                          __bf16* __restrict__ ov) {
    const size_t N = (size_t)HID * HD;
    size_t idx = (size_t)blockIdx.x * 256 + threadIdx.x;   // 3*N total
    size_t m = idx / N;
    size_t r = idx - m * N;
    const float* w = (m == 0) ? wq : (m == 1) ? wk : wv;
    __bf16*      o = (m == 0) ? oq : (m == 1) ? ok : ov;
    int h = (int)(r / HD);
    int d = (int)(r - (size_t)h * HD);
    o[(size_t)d * HID + h] = (__bf16)w[r];
}

// ---------------------------------------------------------------------------
// Kernel 2: projections q/k/v = x @ W (+ xPos on q,k). 4 waves/block, each
// wave owns 64 output columns; A-fragment of x shared by the 3 GEMMs.
// Outputs: q,k row-major bf16 ; v transposed bf16 (HEAD x L per batch).
// ---------------------------------------------------------------------------
__global__ void proj_kernel(const float*  __restrict__ x,
                            const __bf16* __restrict__ wTq,
                            const __bf16* __restrict__ wTk,
                            const __bf16* __restrict__ wTv,
                            __bf16* __restrict__ qo,
                            __bf16* __restrict__ ko,
                            __bf16* __restrict__ vT) {
    const int lane  = threadIdx.x & 31;
    const int cg    = threadIdx.x >> 5;      // col group 0..3 (64 cols each)
    const int rt    = blockIdx.x;            // row tile 0..1023
    const int col16 = lane & 15;
    const int hiSel = lane >> 4;             // 0/1
    const int sel8  = hiSel * 8;
    const int sel16 = hiSel * 16;
    const int g0    = rt * 16;               // global row base (over B*L)
    const int b     = g0 >> 12;              // batch
    const int pos0  = g0 & (LSEQ - 1);       // position base within batch

    v8f aq[4], ak[4], av[4];
#pragma unroll
    for (int t = 0; t < 4; ++t) { aq[t] = {}; ak[t] = {}; av[t] = {}; }

    const float* xrow = x + (size_t)(g0 + col16) * HID;

    for (int kbp = 0; kbp < HID; kbp += 32) {
        float4 f0 = *(const float4*)(xrow + kbp + sel16);
        float4 f1 = *(const float4*)(xrow + kbp + sel16 + 4);
        float4 f2 = *(const float4*)(xrow + kbp + sel16 + 8);
        float4 f3 = *(const float4*)(xrow + kbp + sel16 + 12);
        v16bf a;
        a[0]  = (__bf16)f0.x; a[1]  = (__bf16)f0.y; a[2]  = (__bf16)f0.z; a[3]  = (__bf16)f0.w;
        a[4]  = (__bf16)f1.x; a[5]  = (__bf16)f1.y; a[6]  = (__bf16)f1.z; a[7]  = (__bf16)f1.w;
        a[8]  = (__bf16)f2.x; a[9]  = (__bf16)f2.y; a[10] = (__bf16)f2.z; a[11] = (__bf16)f2.w;
        a[12] = (__bf16)f3.x; a[13] = (__bf16)f3.y; a[14] = (__bf16)f3.z; a[15] = (__bf16)f3.w;
#pragma unroll
        for (int t = 0; t < 4; ++t) {
            int c = cg * 64 + t * 16 + col16;
            size_t wof = (size_t)c * HID + kbp + sel16;
            v16bf bq = *(const v16bf*)(wTq + wof);
            v16bf bk = *(const v16bf*)(wTk + wof);
            v16bf bv = *(const v16bf*)(wTv + wof);
            aq[t] = wmma_bf16(a, bq, aq[t]);
            ak[t] = wmma_bf16(a, bk, ak[t]);
            av[t] = wmma_bf16(a, bv, av[t]);
        }
    }

    // xPos rotary (rotate-half partner lives in lane^1) + bf16 stores.
#pragma unroll
    for (int t = 0; t < 4; ++t) {
        int c  = cg * 64 + t * 16 + col16;
        int hf = c >> 1;
        float ssv  = (2.0f * (float)hf + 0.4f * (float)HD) / (1.4f * (float)HD);
        float lss  = __logf(ssv) * (1.0f / 512.0f);
        float invf = __expf(-(float)hf * (1.0f / 128.0f) * 9.210340371976184f);
        float sgn  = (c & 1) ? 1.0f : -1.0f;
#pragma unroll
        for (int j = 0; j < 8; ++j) {
            int pos = pos0 + j + sel8;
            float sc  = __expf(lss * (float)pos);
            float isc = 1.0f / sc;
            float ang = (float)pos * invf;
            float sv = __sinf(ang), cv = __cosf(ang);

            float xq  = aq[t][j];
            float xqn = __shfl_xor(xq, 1, 32);
            float qv  = xq * (cv * sc) + sgn * xqn * (sv * sc);

            float xk  = ak[t][j];
            float xkn = __shfl_xor(xk, 1, 32);
            float kv  = xk * (cv * isc) + sgn * xkn * (sv * isc);

            size_t row = (size_t)b * LSEQ + pos;
            qo[row * HD + c] = (__bf16)qv;
            ko[row * HD + c] = (__bf16)kv;
            vT[((size_t)b * HD + c) * LSEQ + pos] = (__bf16)av[t][j];
        }
    }
}

// ---------------------------------------------------------------------------
// Kernel 3: out = (q k^T * gamma^(n-m) causal) @ v.
// 4 waves/block = 64 query rows; the 32x256 bf16 k-chunk for each step is
// TDM-loaded into LDS (double buffered, s_wait_tensorcnt) and shared by all
// 4 waves. v streams through global_load_b128 (+ prefetch). S goes through
// a per-wave LDS buffer to switch accumulator layout -> A-fragment layout.
// ---------------------------------------------------------------------------
#define KCHUNK_BYTES (32 * HD * 2)     // 16 KiB per k buffer

__global__ void __launch_bounds__(128) ret_kernel(const __bf16* __restrict__ q,
                                                  const __bf16* __restrict__ k,
                                                  const __bf16* __restrict__ vT,
                                                  float* __restrict__ out) {
    // kbuf is first shared object -> LDS byte offset 0 (TDM targets it).
    __shared__ __attribute__((aligned(128))) __bf16 kbuf[2][32 * HD];
    __shared__ __attribute__((aligned(16)))  __bf16 sbuf[4][16 * 32];

    const int lane  = threadIdx.x & 31;
    const int wave  = threadIdx.x >> 5;       // 0..3
    const int col16 = lane & 15;
    const int hiSel = lane >> 4;
    const int sel8  = hiSel * 8;
    const int sel16 = hiSel * 16;
    const int nb    = blockIdx.x & 63;        // 64-row group within batch
    const int b     = blockIdx.x >> 6;        // batch
    const int n0    = nb * 4 + wave;          // this wave's 16-row q tile
    const int smax  = 2 * nb + 1;             // 32-key steps cover [0, 64nb+64)

    // Kick off TDM for step 0 while q fragments load.
    if (wave == 0) {
        tdm_load_tile_2d(k + (size_t)b * LSEQ * HD, 0u,
                         /*tile*/ HD, 32, /*tensor*/ HD, LSEQ, /*stride0*/ HD);
    }

    // q A-fragments for all of K = HEAD = 256 (8 chunks of 32)
    const __bf16* qrow = q + ((size_t)b * LSEQ + n0 * 16 + col16) * HD;
    v16bf qf[8];
#pragma unroll
    for (int kc = 0; kc < 8; ++kc) {
        const __bf16* p = qrow + kc * 32 + sel16;
        v8bf lo = *(const v8bf*)p;
        v8bf hi = *(const v8bf*)(p + 8);
        v16bf a;
#pragma unroll
        for (int i = 0; i < 8; ++i) { a[i] = lo[i]; a[i + 8] = hi[i]; }
        qf[kc] = a;
    }

    v8f Oacc[16];
#pragma unroll
    for (int t = 0; t < 16; ++t) Oacc[t] = {};

    const float ln_gamma = -0.0317486983145803f;   // ln(0.96875)

    for (int s = 0; s <= smax; ++s) {
        const int cur = s & 1;
        if (wave == 0) {
            if (s < smax) {   // overlap: DMA chunk s+1 while computing s
                tdm_load_tile_2d(k + ((size_t)b * LSEQ + (size_t)(s + 1) * 32) * HD,
                                 (unsigned)((s + 1) & 1) * KCHUNK_BYTES,
                                 HD, 32, HD, LSEQ, HD);
                __builtin_amdgcn_s_wait_tensorcnt(1);  // chunk s resident
            } else {
                __builtin_amdgcn_s_wait_tensorcnt(0);
            }
        }
        __syncthreads();

        if (s < smax) {   // prefetch next v chunk (global_prefetch_b8)
            __builtin_prefetch(vT + ((size_t)b * HD + lane * 8) * LSEQ + (s + 1) * 32, 0, 3);
        }

        // ---- S = q . k^T for 32 keys (two 16x16 tiles) from LDS k-chunk ----
#pragma unroll
        for (int t = 0; t < 2; ++t) {
            v8f Sacc = {};
            const __bf16* krow = &kbuf[cur][(t * 16 + col16) * HD + sel16];
#pragma unroll
            for (int kc = 0; kc < 8; ++kc) {
                v16bf bf = *(const v16bf*)(krow + kc * 32);
                Sacc = wmma_bf16(qf[kc], bf, Sacc);
            }
            int m = s * 32 + t * 16 + col16;  // global key index
#pragma unroll
            for (int j = 0; j < 8; ++j) {
                int n = n0 * 16 + j + sel8;
                int diff = n - m;
                float dec = (diff >= 0) ? __expf((float)diff * ln_gamma) : 0.0f;
                sbuf[wave][(j + sel8) * 32 + t * 16 + col16] = (__bf16)(Sacc[j] * dec);
            }
        }

        // ---- reload S as an A-fragment (16x32, K = key index) ----
        const __bf16* sp = &sbuf[wave][col16 * 32 + sel16];
        v8bf lo = *(const v8bf*)sp;
        v8bf hi = *(const v8bf*)(sp + 8);
        v16bf sf;
#pragma unroll
        for (int i = 0; i < 8; ++i) { sf[i] = lo[i]; sf[i + 8] = hi[i]; }

        // ---- O += S @ v  (v transposed: B column = head dim) ----
#pragma unroll
        for (int t2 = 0; t2 < 16; ++t2) {
            const __bf16* vp = vT + ((size_t)b * HD + t2 * 16 + col16) * LSEQ
                                  + s * 32 + sel16;
            v16bf bf = *(const v16bf*)vp;
            Oacc[t2] = wmma_bf16(sf, bf, Oacc[t2]);
        }
        __syncthreads();   // kbuf[cur] free before TDM (step s+2) rewrites it
    }

    // ---- store output f32 ----
#pragma unroll
    for (int t2 = 0; t2 < 16; ++t2) {
        int c = t2 * 16 + col16;
#pragma unroll
        for (int j = 0; j < 8; ++j) {
            int n = n0 * 16 + j + sel8;
            out[((size_t)b * LSEQ + n) * HD + c] = Oacc[t2][j];
        }
    }
}

// ---------------------------------------------------------------------------
extern "C" void kernel_launch(void* const* d_in, const int* in_sizes, int n_in,
                              void* d_out, int out_size, void* d_ws, size_t ws_size,
                              hipStream_t stream) {
    const float* x  = (const float*)d_in[0];
    const float* wq = (const float*)d_in[1];
    const float* wk = (const float*)d_in[2];
    const float* wv = (const float*)d_in[3];
    float* out = (float*)d_out;

    // Workspace (bf16): wT q/k/v = 3 MiB ; q | k | vT = 24 MiB
    __bf16* wTq = (__bf16*)d_ws;
    __bf16* wTk = wTq + (size_t)HD * HID;
    __bf16* wTv = wTk + (size_t)HD * HID;
    __bf16* qb  = wTv + (size_t)HD * HID;
    __bf16* kb  = qb  + (size_t)BATCH * LSEQ * HD;
    __bf16* vT  = kb  + (size_t)BATCH * LSEQ * HD;

    wt_kernel<<<dim3(3 * HID), 256, 0, stream>>>(wq, wk, wv, wTq, wTk, wTv);

    proj_kernel<<<dim3((BATCH * LSEQ) / 16), 128, 0, stream>>>(
        x, wTq, wTk, wTv, qb, kb, vT);

    // 1024 blocks x 128 threads: 4 q-tiles (64 rows) share each TDM k-chunk
    ret_kernel<<<dim3(BATCH * (LSEQ / 16)), 128, 0, stream>>>(qb, kb, vT, out);
}
